// TransitionGraphEncoder_8727373545808
// MI455X (gfx1250) — matrix operand order repaired and checked by reading, not compiled, not run on hardware
//
#include <hip/hip_runtime.h>
#include <hip/hip_bf16.h>

typedef __attribute__((ext_vector_type(16))) _Float16 v16h;
typedef __attribute__((ext_vector_type(8)))  float    v8f;

union FragU {
    v16h v;
    _Float16 h[16];
};

// ---------------------------------------------------------------------------
// WMMA fragment helpers (CDNA5 16x16x32 f16, wave32)
// A layout (16x32, lane = M): VGPR p<4 : k = grp*8 + 2p   (+0,+1)
//                             VGPR p>=4: k = 16 + grp*8 + 2(p-4)
// B layout (32x16, lane = N): packed in the same k order as A,
//                             VGPR p : k = a_k0(grp, p)   (+0,+1)
// C layout (16x16 f32): vgpr r holds row M = grp*8 + r, N = lane&15
// ---------------------------------------------------------------------------
__device__ inline int a_k0(int grp, int p) {
    return ((p & 4) << 2) + grp * 8 + ((p & 3) << 1);
}

__device__ inline v16h load_a_frag_f16(const _Float16* row, int grp) {
    FragU f;
#pragma unroll
    for (int p = 0; p < 8; ++p) {
        int k0 = a_k0(grp, p);
        f.h[2 * p]     = row[k0];
        f.h[2 * p + 1] = row[k0 + 1];
    }
    return f.v;
}

__device__ inline v16h load_a_frag_f32(const float* row, int grp) {
    FragU f;
#pragma unroll
    for (int p = 0; p < 8; ++p) {
        int k0 = a_k0(grp, p);
        f.h[2 * p]     = (_Float16)row[k0];
        f.h[2 * p + 1] = (_Float16)row[k0 + 1];
    }
    return f.v;
}

__device__ inline v16h load_b_frag(const _Float16* packed, int frag, int lane) {
    return *(const v16h*)(packed + ((size_t)(frag * 32 + lane)) * 16);
}

// order-preserving float <-> uint encoding for atomic max
__device__ inline unsigned fenc(float x) {
    unsigned u = __float_as_uint(x);
    return (u & 0x80000000u) ? ~u : (u | 0x80000000u);
}
__device__ inline float fdec(unsigned u) {
    return (u & 0x80000000u) ? __uint_as_float(u ^ 0x80000000u) : __uint_as_float(~u);
}

// ---------------------------------------------------------------------------
// Pack weights into f16 B-fragments once.
// pack layout (halfs): w1p @0 (32 frags), w2p @16384 (8), wm1p @20480 (24),
//                      wm2p @32768 (8), rel_f16 @36864 (448)
// ---------------------------------------------------------------------------
__global__ void prep_kernel(const float* __restrict__ W1, const float* __restrict__ W2,
                            const float* __restrict__ Wm1, const float* __restrict__ Wm2,
                            const float* __restrict__ relEmb, _Float16* __restrict__ pack) {
    const int total = (32 + 8 + 24 + 8) * 32;   // frag-lane units
    for (int idx = blockIdx.x * blockDim.x + threadIdx.x; idx < total + 448;
         idx += gridDim.x * blockDim.x) {
        if (idx < total) {
            int u = idx;
            const float* W;
            _Float16* dstbase;
            if (u < 1024)      { W = W1;  dstbase = pack;         }
            else if (u < 1280) { u -= 1024; W = W2;  dstbase = pack + 16384; }
            else if (u < 2048) { u -= 1280; W = Wm1; dstbase = pack + 20480; }
            else               { u -= 2048; W = Wm2; dstbase = pack + 32768; }
            int fragIdx = u >> 5, lane = u & 31;
            int kt = fragIdx >> 2, nt = fragIdx & 3;
            int grp = lane >> 4, n = lane & 15;
            _Float16* d = dstbase + (size_t)u * 16;
#pragma unroll
            for (int p = 0; p < 8; ++p) {
                int k0 = kt * 32 + ((p & 4) << 2) + grp * 8 + ((p & 3) << 1);
                d[2 * p]     = (_Float16)W[(size_t)k0 * 64 + nt * 16 + n];
                d[2 * p + 1] = (_Float16)W[(size_t)(k0 + 1) * 64 + nt * 16 + n];
            }
        } else {
            int j = idx - total;
            (pack + 36864)[j] = (_Float16)relEmb[j];
        }
    }
}

// ---------------------------------------------------------------------------
// Node MLP: h = relu( relu(LN(X@W1+b1))@W2 + b2 )   (2 waves / block, 16 nodes / wave)
// ---------------------------------------------------------------------------
__global__ void node_mlp_kernel(const float* __restrict__ X,
                                const float* __restrict__ b1, const float* __restrict__ g1,
                                const float* __restrict__ be1, const float* __restrict__ b2,
                                const _Float16* __restrict__ w1p, const _Float16* __restrict__ w2p,
                                float* __restrict__ hOut, _Float16* __restrict__ hHalf, int Nn) {
    __shared__ _Float16 ldsT[2][16 * 64];
    const int wave = threadIdx.x >> 5;
    const int lane = threadIdx.x & 31;
    const int grp = lane >> 4, lm = lane & 15;
    const int nodeBase = (blockIdx.x * 2 + wave) * 16;
    int myNode = nodeBase + lm;
    if (myNode >= Nn) myNode = Nn - 1;
    const float* row = X + (size_t)myNode * 256;

    float b1n[4], g1n[4], be1n[4], b2n[4];
#pragma unroll
    for (int nt = 0; nt < 4; ++nt) {
        int n = nt * 16 + lm;
        b1n[nt] = b1[n]; g1n[nt] = g1[n]; be1n[nt] = be1[n]; b2n[nt] = b2[n];
    }

    v8f acc[4];
#pragma unroll
    for (int nt = 0; nt < 4; ++nt)
#pragma unroll
        for (int i = 0; i < 8; ++i) acc[nt][i] = 0.f;

    for (int kt = 0; kt < 8; ++kt) {
        v16h a = load_a_frag_f32(row + kt * 32, grp);
#pragma unroll
        for (int nt = 0; nt < 4; ++nt) {
            v16h b = load_b_frag(w1p, (kt << 2) + nt, lane);
            acc[nt] = __builtin_amdgcn_wmma_f32_16x16x32_f16(
                false, a, false, b, (short)0, acc[nt], false, false);
        }
    }

    // bias + layernorm + relu, write transposed f16 tile to LDS
#pragma unroll
    for (int r = 0; r < 8; ++r) {
        float x[4], s = 0.f, ss = 0.f;
#pragma unroll
        for (int nt = 0; nt < 4; ++nt) {
            x[nt] = acc[nt][r] + b1n[nt];
            s += x[nt];
            ss += x[nt] * x[nt];
        }
        for (int m = 1; m <= 8; m <<= 1) {
            s  += __shfl_xor(s, m, 32);
            ss += __shfl_xor(ss, m, 32);
        }
        float mu = s * (1.f / 64.f);
        float var = ss * (1.f / 64.f) - mu * mu;
        float rs = rsqrtf(var + 1e-5f);
        int mrow = grp * 8 + r;
#pragma unroll
        for (int nt = 0; nt < 4; ++nt) {
            float y = fmaxf((x[nt] - mu) * rs * g1n[nt] + be1n[nt], 0.f);
            ldsT[wave][mrow * 64 + nt * 16 + lm] = (_Float16)y;
        }
    }
    __syncthreads();

    // second layer: 16x64 @ 64x64
    v8f acc2[4];
#pragma unroll
    for (int nt = 0; nt < 4; ++nt)
#pragma unroll
        for (int i = 0; i < 8; ++i) acc2[nt][i] = 0.f;
    const _Float16* trow = &ldsT[wave][lm * 64];
#pragma unroll
    for (int kt = 0; kt < 2; ++kt) {
        v16h a = load_a_frag_f16(trow + kt * 32, grp);
#pragma unroll
        for (int nt = 0; nt < 4; ++nt) {
            v16h b = load_b_frag(w2p, (kt << 2) + nt, lane);
            acc2[nt] = __builtin_amdgcn_wmma_f32_16x16x32_f16(
                false, a, false, b, (short)0, acc2[nt], false, false);
        }
    }
#pragma unroll
    for (int r = 0; r < 8; ++r) {
        int node = nodeBase + grp * 8 + r;
        if (node < Nn) {
#pragma unroll
            for (int nt = 0; nt < 4; ++nt) {
                float y = fmaxf(acc2[nt][r] + b2n[nt], 0.f);
                hOut[(size_t)node * 64 + nt * 16 + lm] = y;
                hHalf[(size_t)node * 64 + nt * 16 + lm] = (_Float16)y;
            }
        }
    }
}

// ---------------------------------------------------------------------------
// Edge MLP + scatter-add: msg = relu([h_s|h_d|rel|w]@Wm1+bm1)@Wm2+bm2 ; agg[dst]+=msg
// 4 waves / block, 16 edges / wave.  K=192 via WMMA + rank-1 update for w.
// ---------------------------------------------------------------------------
__global__ void edge_mlp_kernel(const float* __restrict__ TE, const _Float16* __restrict__ hHalf,
                                const _Float16* __restrict__ relh,
                                const float* __restrict__ Wm1, const float* __restrict__ bm1,
                                const float* __restrict__ bm2,
                                const _Float16* __restrict__ wm1p, const _Float16* __restrict__ wm2p,
                                float* __restrict__ agg) {
    __shared__ _Float16 ldsT[4][16 * 64];
    const int wave = threadIdx.x >> 5;
    const int lane = threadIdx.x & 31;
    const int grp = lane >> 4, lm = lane & 15;
    const int tile = blockIdx.x * 4 + wave;
    const int e = tile * 16 + lm;

    float4 te = ((const float4*)TE)[e];
    int src = (int)te.x, dst = (int)te.y, rel = (int)te.z;
    float w = te.w;
    const _Float16* hs = hHalf + (size_t)src * 64;
    const _Float16* hd = hHalf + (size_t)dst * 64;
    const _Float16* re = relh + rel * 64;

    float bm1n[4], wlast[4], bm2n[4];
    const float* wm1last = Wm1 + 192 * 64;
#pragma unroll
    for (int nt = 0; nt < 4; ++nt) {
        int n = nt * 16 + lm;
        bm1n[nt] = bm1[n]; wlast[nt] = wm1last[n]; bm2n[nt] = bm2[n];
    }

    v8f acc[4];
#pragma unroll
    for (int nt = 0; nt < 4; ++nt)
#pragma unroll
        for (int i = 0; i < 8; ++i) acc[nt][i] = 0.f;

#pragma unroll
    for (int kt = 0; kt < 6; ++kt) {
        const _Float16* rowp = (kt < 2) ? hs : (kt < 4) ? hd : re;
        v16h a = load_a_frag_f16(rowp + (kt & 1) * 32, grp);
#pragma unroll
        for (int nt = 0; nt < 4; ++nt) {
            v16h b = load_b_frag(wm1p, (kt << 2) + nt, lane);
            acc[nt] = __builtin_amdgcn_wmma_f32_16x16x32_f16(
                false, a, false, b, (short)0, acc[nt], false, false);
        }
    }

    // layer-1 epilogue: + bm1 + w*Wm1[192,:], relu, transpose via LDS
#pragma unroll
    for (int r = 0; r < 8; ++r) {
        int mrow = grp * 8 + r;
        float wm = __shfl(w, mrow, 32);
#pragma unroll
        for (int nt = 0; nt < 4; ++nt) {
            float x = acc[nt][r] + bm1n[nt] + wm * wlast[nt];
            ldsT[wave][mrow * 64 + nt * 16 + lm] = (_Float16)fmaxf(x, 0.f);
        }
    }
    __syncthreads();

    v8f acc2[4];
#pragma unroll
    for (int nt = 0; nt < 4; ++nt)
#pragma unroll
        for (int i = 0; i < 8; ++i) acc2[nt][i] = 0.f;
    const _Float16* trow = &ldsT[wave][lm * 64];
#pragma unroll
    for (int kt = 0; kt < 2; ++kt) {
        v16h a = load_a_frag_f16(trow + kt * 32, grp);
#pragma unroll
        for (int nt = 0; nt < 4; ++nt) {
            v16h b = load_b_frag(wm2p, (kt << 2) + nt, lane);
            acc2[nt] = __builtin_amdgcn_wmma_f32_16x16x32_f16(
                false, a, false, b, (short)0, acc2[nt], false, false);
        }
    }

    // scatter-add into agg[dst]
#pragma unroll
    for (int r = 0; r < 8; ++r) {
        int mrow = grp * 8 + r;
        int dd = __shfl(dst, mrow, 32);
#pragma unroll
        for (int nt = 0; nt < 4; ++nt) {
            atomicAdd(&agg[(size_t)dd * 64 + nt * 16 + lm], acc2[nt][r] + bm2n[nt]);
        }
    }
}

// ---------------------------------------------------------------------------
// nodes = LN(h + agg); accumulate column sum + column max (order-encoded uint)
// ---------------------------------------------------------------------------
__global__ void node_reduce_kernel(const float* __restrict__ h, const float* __restrict__ agg,
                                   const float* __restrict__ gn, const float* __restrict__ bn,
                                   float* __restrict__ sumg, unsigned* __restrict__ maxg, int Nn) {
    __shared__ float sAcc[64];
    __shared__ unsigned uAcc[64];
    const int lane = threadIdx.x & 31;
    const int wave = threadIdx.x >> 5;
    if (threadIdx.x < 64) { sAcc[threadIdx.x] = 0.f; uAcc[threadIdx.x] = 0u; }
    __syncthreads();

    float g0 = gn[lane], g1v = gn[32 + lane];
    float b0 = bn[lane], b1v = bn[32 + lane];
    const int gw = blockIdx.x * 8 + wave;
    const int nw = gridDim.x * 8;
    float ls0 = 0.f, ls1 = 0.f;
    unsigned lm0 = 0u, lm1 = 0u;
    for (int r = gw; r < Nn; r += nw) {
        size_t base = (size_t)r * 64;
        float x0 = h[base + lane] + agg[base + lane];
        float x1 = h[base + 32 + lane] + agg[base + 32 + lane];
        float s = x0 + x1, ss = x0 * x0 + x1 * x1;
        for (int m = 1; m <= 16; m <<= 1) {
            s  += __shfl_xor(s, m, 32);
            ss += __shfl_xor(ss, m, 32);
        }
        float mu = s * (1.f / 64.f);
        float var = ss * (1.f / 64.f) - mu * mu;
        float rs = rsqrtf(var + 1e-5f);
        float y0 = (x0 - mu) * rs * g0 + b0;
        float y1 = (x1 - mu) * rs * g1v + b1v;
        ls0 += y0; ls1 += y1;
        unsigned e0 = fenc(y0), e1 = fenc(y1);
        lm0 = lm0 > e0 ? lm0 : e0;
        lm1 = lm1 > e1 ? lm1 : e1;
    }
    atomicAdd(&sAcc[lane], ls0);
    atomicAdd(&sAcc[32 + lane], ls1);
    atomicMax(&uAcc[lane], lm0);
    atomicMax(&uAcc[32 + lane], lm1);
    __syncthreads();
    if (threadIdx.x < 64) {
        atomicAdd(&sumg[threadIdx.x], sAcc[threadIdx.x]);
        atomicMax(&maxg[threadIdx.x], uAcc[threadIdx.x]);
    }
}

// ---------------------------------------------------------------------------
// head: g = [mean | max] (128) ; out = relu(g@Wg1+bg1)@Wg2+bg2  (64)
// ---------------------------------------------------------------------------
__global__ void head_kernel(const float* __restrict__ sumg, const unsigned* __restrict__ maxg,
                            const float* __restrict__ Wg1, const float* __restrict__ bg1,
                            const float* __restrict__ Wg2, const float* __restrict__ bg2,
                            float* __restrict__ out, int Nn) {
    __shared__ float gvec[128];
    __shared__ float tvec[64];
    int t = threadIdx.x;   // 64 threads
    gvec[t]      = sumg[t] / (float)Nn;
    gvec[64 + t] = fdec(maxg[t]);
    __syncthreads();
    float a = bg1[t];
    for (int j = 0; j < 128; ++j) a += gvec[j] * Wg1[j * 64 + t];
    tvec[t] = fmaxf(a, 0.f);
    __syncthreads();
    float o = bg2[t];
    for (int j = 0; j < 64; ++j) o += tvec[j] * Wg2[j * 64 + t];
    out[t] = o;
}

// ---------------------------------------------------------------------------
extern "C" void kernel_launch(void* const* d_in, const int* in_sizes, int n_in,
                              void* d_out, int out_size, void* d_ws, size_t ws_size,
                              hipStream_t stream) {
    const float* X    = (const float*)d_in[0];
    const float* TE   = (const float*)d_in[1];
    const float* W1   = (const float*)d_in[2];
    const float* b1   = (const float*)d_in[3];
    const float* g1   = (const float*)d_in[4];
    const float* be1  = (const float*)d_in[5];
    const float* W2   = (const float*)d_in[6];
    const float* b2   = (const float*)d_in[7];
    const float* rele = (const float*)d_in[8];
    const float* Wm1  = (const float*)d_in[9];
    const float* bm1  = (const float*)d_in[10];
    const float* Wm2  = (const float*)d_in[11];
    const float* bm2  = (const float*)d_in[12];
    const float* gn   = (const float*)d_in[13];
    const float* bn   = (const float*)d_in[14];
    const float* Wg1  = (const float*)d_in[15];
    const float* bg1  = (const float*)d_in[16];
    const float* Wg2  = (const float*)d_in[17];
    const float* bg2  = (const float*)d_in[18];
    float* out = (float*)d_out;

    const int Nn = 100000;
    char* ws = (char*)d_ws;
    float*    hF   = (float*)ws;                           // 25,600,000 B
    _Float16* hH   = (_Float16*)(ws + 25600000);           // 12,800,000 B
    float*    agg  = (float*)(ws + 38400000);              // 25,600,000 B (zeroed)
    float*    sumg = (float*)(ws + 64000000);              // 256 B (zeroed)
    unsigned* maxg = (unsigned*)(ws + 64000256);           // 256 B (zeroed)
    _Float16* pack = (_Float16*)(ws + 64000512);           // ~75 KB

    hipMemsetAsync(agg, 0, 25600000 + 512, stream);

    prep_kernel<<<16, 256, 0, stream>>>(W1, W2, Wm1, Wm2, rele, pack);

    // 3125 blocks * 2 waves * 16 nodes = 100000
    node_mlp_kernel<<<3125, 64, 0, stream>>>(X, b1, g1, be1, b2,
                                             pack, pack + 16384, hF, hH, Nn);

    // 12500 blocks * 4 waves * 16 edges = 800000
    edge_mlp_kernel<<<12500, 128, 0, stream>>>(TE, hH, pack + 36864, Wm1, bm1, bm2,
                                               pack + 20480, pack + 32768, agg);

    node_reduce_kernel<<<512, 256, 0, stream>>>(hF, agg, gn, bn, sumg, maxg, Nn);

    head_kernel<<<1, 64, 0, stream>>>(sumg, maxg, Wg1, bg1, Wg2, bg2, out, Nn);
}
